// MultiHeadAttention2d_43439299231790
// MI455X (gfx1250) — matrix operand order repaired and checked by reading, not compiled, not run on hardware
//
#include <hip/hip_runtime.h>
#include <hip/hip_bf16.h>

// ---------------------------------------------------------------------------
// MultiHeadAttention2d for MI455X (gfx1250): fused flash-attention with
// v_wmma_f32_16x16x32_f16, async global->LDS staging (ASYNCcnt), f16 Q/K/V
// in workspace, f32 accumulation.
// ---------------------------------------------------------------------------

#define BATCH   4
#define CIN     64
#define NHEADS  4
#define DHEAD   64
#define NTOK    4096          // H*W
#define NPAIR   (NHEADS * BATCH)

typedef __attribute__((ext_vector_type(16))) _Float16     v16h;
typedef __attribute__((ext_vector_type(8)))  float        v8f;
typedef __attribute__((ext_vector_type(8)))  unsigned int v8u;

union F16Frag { v16h h; v8u u; };

// Issue one 16-byte async global->LDS transfer (gfx1250, tracked by ASYNCcnt).
// VDST VGPR carries the LDS byte offset; VADDR the 64-bit global address.
__device__ __forceinline__ void async_copy_b128(const _Float16* gsrc, _Float16* ldst) {
    const unsigned           ldsa = (unsigned)(unsigned long long)(uintptr_t)ldst;
    const unsigned long long ga   = (unsigned long long)(uintptr_t)gsrc;
    asm volatile("global_load_async_to_lds_b128 %0, %1, off"
                 :: "v"(ldsa), "v"(ga) : "memory");
}
__device__ __forceinline__ void async_wait0() {
    asm volatile("s_wait_asynccnt 0x0" ::: "memory");
}

// ---- WMMA fragment loaders -------------------------------------------------
// A-matrix (16x32 f16) lane layout (ISA 7.12.2):
//   lane row = lane&15, half = lane>>4
//   VGPR j holds K = (j/4)*16 + half*8 + (j%4)*2 , +1   (packed pair = 1 dword)
__device__ __forceinline__ v16h load_a_frag(const _Float16* base, int ldm, int lane) {
    const int row  = lane & 15;
    const int half = lane >> 4;
    F16Frag f;
#pragma unroll
    for (int j = 0; j < 8; ++j) {
        const int kk = ((j >> 2) << 4) + (half << 3) + ((j & 3) << 1);
        f.u[j] = *(const unsigned int*)(base + row * ldm + kk);
    }
    return f.h;
}

// B-matrix (32x16 f16): logical B[k][n] = src[n*ldm + k] (K contiguous in src).
//   lane col = lane&15, half = lane>>4 (same K order as A-matrix);
//   VGPR j holds K = (j/4)*16 + half*8 + (j%4)*2 , +1
__device__ __forceinline__ v16h load_bT_frag(const _Float16* base, int ldm, int lane) {
    const int col  = lane & 15;
    const int half = lane >> 4;
    F16Frag f;
#pragma unroll
    for (int j = 0; j < 8; ++j) {
        const int kk = ((j >> 2) << 4) + (half << 3) + ((j & 3) << 1);
        f.u[j] = *(const unsigned int*)(base + col * ldm + kk);
    }
    return f.h;
}

// ---------------------------------------------------------------------------
// Kernel 1: 1x1-conv projections  (B,C,H,W) x (O,C) -> head-major f16 Q/K/V
// layout: Qh[((h*B + b)*NTOK + n)*DHEAD + d]
// ---------------------------------------------------------------------------
__device__ __forceinline__ void proj_one(const float xs[CIN][64], const float* __restrict__ w,
                                         float bias, _Float16* __restrict__ dst) {
    float acc[64];
#pragma unroll
    for (int p = 0; p < 64; ++p) acc[p] = bias;
    for (int c = 0; c < CIN; ++c) {
        const float wv = w[c];
#pragma unroll
        for (int p = 0; p < 64; ++p) acc[p] += wv * xs[c][p];
    }
    for (int p = 0; p < 64; ++p) dst[(long)p * DHEAD] = (_Float16)acc[p];
}

__global__ __launch_bounds__(256) void proj_kernel(
    const float* __restrict__ x,  const float* __restrict__ y,
    const float* __restrict__ Wq, const float* __restrict__ bq,
    const float* __restrict__ Wk, const float* __restrict__ bk,
    const float* __restrict__ Wv, const float* __restrict__ bv,
    _Float16* __restrict__ Qh, _Float16* __restrict__ Kh, _Float16* __restrict__ Vh) {

    __shared__ float xs[CIN][64];

    const int tid = threadIdx.x;           // 0..255 == output channel o
    const int blk = blockIdx.x;            // 0..(B*NTOK/64 - 1)
    const int b   = blk >> 6;              // 64 pixel-tiles per batch image
    const int n0  = (blk & 63) * 64;
    const int o   = tid;
    const int hh  = o >> 6;
    const int dd  = o & 63;
    const long pair = (long)hh * BATCH + b;
    const long obase = (pair * NTOK + n0) * DHEAD + dd;

    // pass 1: x -> Q   (vectorized: 4 floats per load, 4 loads per thread)
    for (int idx = tid; idx < CIN * 16; idx += 256) {
        const int c = idx >> 4, p = (idx & 15) * 4;
        *(float4*)&xs[c][p] = *(const float4*)&x[((long)(b * CIN + c)) * NTOK + n0 + p];
    }
    __syncthreads();
    proj_one(xs, Wq + o * CIN, bq[o], Qh + obase);
    __syncthreads();

    // pass 2: y -> K, V
    for (int idx = tid; idx < CIN * 16; idx += 256) {
        const int c = idx >> 4, p = (idx & 15) * 4;
        *(float4*)&xs[c][p] = *(const float4*)&y[((long)(b * CIN + c)) * NTOK + n0 + p];
    }
    __syncthreads();
    proj_one(xs, Wk + o * CIN, bk[o], Kh + obase);
    proj_one(xs, Wv + o * CIN, bv[o], Vh + obase);
}

// ---------------------------------------------------------------------------
// Kernel 2: fused flash attention, one WG per (pair, 64-row query tile).
// 256 threads = 8 waves; wave w owns query row-block (w&3)*16 and the
// 32-column half (w>>2)*32 of every 64x64 tile (S and O alike).
// ---------------------------------------------------------------------------
__global__ __launch_bounds__(256) void attn_kernel(
    const _Float16* __restrict__ Qh, const _Float16* __restrict__ Kh,
    const _Float16* __restrict__ Vh, float* __restrict__ out) {

    __shared__ _Float16 qlds[64 * DHEAD];   // Q tile, row-major (n x d)
    __shared__ _Float16 klds[64 * DHEAD];   // K tile, row-major (n' x d)
    __shared__ _Float16 vtlds[DHEAD * 64];  // V tile transposed (d x kk)
    __shared__ _Float16 plds[64 * 64];      // P = exp(S - m), row-major (n x kk)
    __shared__ float    slds[64 * 64];      // raw scores f32
    __shared__ float    row_m[64], row_l[64], row_c[64];

    const int tid   = threadIdx.x;
    const int lane  = tid & 31;
    const int wave  = tid >> 5;
    const int rblk  = wave & 3;             // query row block (16 rows)
    const int chalf = wave >> 2;            // column half (2 tiles of 16)
    const int qtile = blockIdx.x;           // 0..63
    const int pair  = blockIdx.y;           // 0..15 (= h*B + b)
    const int h     = pair >> 2;
    const int b     = pair & 3;

    const long pbase = (long)pair * NTOK * DHEAD;
    const _Float16* qg = Qh + pbase + (long)qtile * 64 * DHEAD;

    // Q tile -> LDS via async DMA (2 x 16B per thread covers 8 KB)
#pragma unroll
    for (int c = 0; c < 2; ++c) {
        const int off = (tid + c * 256) * 8;        // in halves (16B chunks)
        async_copy_b128(qg + off, &qlds[off]);
    }
    if (tid < 64) { row_m[tid] = -1.0e30f; row_l[tid] = 0.0f; }
    async_wait0();
    __syncthreads();

    // Preload this wave's Q A-fragments (rows rblk*16.., k-chunks d:0-31 / 32-63)
    v16h qfrag[2];
    qfrag[0] = load_a_frag(&qlds[(rblk * 16) * DHEAD + 0],  DHEAD, lane);
    qfrag[1] = load_a_frag(&qlds[(rblk * 16) * DHEAD + 32], DHEAD, lane);

    v8f oacc[2];
#pragma unroll
    for (int t = 0; t < 2; ++t)
#pragma unroll
        for (int i = 0; i < 8; ++i) oacc[t][i] = 0.0f;

    for (int j = 0; j < NTOK / 64; ++j) {
        const _Float16* kg = Kh + pbase + (long)j * 64 * DHEAD;
        const _Float16* vg = Vh + pbase + (long)j * 64 * DHEAD;
        if (j + 1 < NTOK / 64) {            // hint next tiles toward near caches
            __builtin_prefetch(kg + 64 * DHEAD, 0, 3);
            __builtin_prefetch(vg + 64 * DHEAD, 0, 3);
        }

        // K tile: straight async DMA into LDS (row-major, 8 KB)
#pragma unroll
        for (int c = 0; c < 2; ++c) {
            const int off = (tid + c * 256) * 8;
            async_copy_b128(kg + off, &klds[off]);
        }
        // V tile: vectorized load + transposed scatter into LDS
        union { uint4 u4; _Float16 hx[8]; } vv;
#pragma unroll
        for (int c = 0; c < 2; ++c) {
            const int chunk = tid + c * 256;        // 0..511
            const int kk  = chunk >> 3;
            const int dd0 = (chunk & 7) * 8;
            vv.u4 = *(const uint4*)(vg + kk * DHEAD + dd0);
#pragma unroll
            for (int i = 0; i < 8; ++i) vtlds[(dd0 + i) * 64 + kk] = vv.hx[i];
        }
        async_wait0();
        __syncthreads();

        // ---- S = Q * K^T (this wave's 16x32 strip) ----
#pragma unroll
        for (int t = 0; t < 2; ++t) {
            const int cb = chalf * 2 + t;
            v8f s;
#pragma unroll
            for (int i = 0; i < 8; ++i) s[i] = 0.0f;
#pragma unroll
            for (int ks = 0; ks < 2; ++ks) {
                const v16h bf = load_bT_frag(&klds[(cb * 16) * DHEAD + ks * 32], DHEAD, lane);
                s = __builtin_amdgcn_wmma_f32_16x16x32_f16(
                        false, qfrag[ks], false, bf, (short)0, s, false, false);
            }
            const int col = cb * 16 + (lane & 15);
#pragma unroll
            for (int i = 0; i < 8; ++i) {
                const int row = rblk * 16 + ((lane >> 4) << 3) + i;
                slds[row * 64 + col] = s[i];
            }
        }
        __syncthreads();

        // ---- online softmax (one thread per query row) ----
        if (tid < 64) {
            const int r = tid;
            const float m_old = row_m[r];
            float mx = m_old;
            for (int c2 = 0; c2 < 64; ++c2) mx = fmaxf(mx, slds[r * 64 + c2]);
            const float corr = __expf(m_old - mx);
            float sum = 0.0f;
            for (int c2 = 0; c2 < 64; ++c2) {
                const float p = __expf(slds[r * 64 + c2] - mx);
                sum += p;
                plds[r * 64 + c2] = (_Float16)p;
            }
            row_m[r] = mx;
            row_l[r] = row_l[r] * corr + sum;
            row_c[r] = corr;
        }
        __syncthreads();

        // ---- rescale accumulators, then O += P * V ----
        {
            const int rb0 = rblk * 16 + ((lane >> 4) << 3);
#pragma unroll
            for (int i = 0; i < 8; ++i) {
                const float cf = row_c[rb0 + i];
                oacc[0][i] *= cf;
                oacc[1][i] *= cf;
            }
        }
#pragma unroll
        for (int ks = 0; ks < 2; ++ks) {
            const v16h pf = load_a_frag(&plds[(rblk * 16) * 64 + ks * 32], 64, lane);
#pragma unroll
            for (int t = 0; t < 2; ++t) {
                const int dt = chalf * 2 + t;
                const v16h vf = load_bT_frag(&vtlds[(dt * 16) * 64 + ks * 32], 64, lane);
                oacc[t] = __builtin_amdgcn_wmma_f32_16x16x32_f16(
                              false, pf, false, vf, (short)0, oacc[t], false, false);
            }
        }
        __syncthreads();
    }

    // ---- finalize: divide by row sums, write out[b, h, d, n] ----
    const int rb0 = rblk * 16 + ((lane >> 4) << 3);
#pragma unroll
    for (int i = 0; i < 8; ++i) {
        const float inv = 1.0f / row_l[rb0 + i];
        const int n = qtile * 64 + rb0 + i;
#pragma unroll
        for (int t = 0; t < 2; ++t) {
            const int dd = (chalf * 2 + t) * 16 + (lane & 15);
            out[((long)((b * NHEADS + h) * DHEAD + dd)) * NTOK + n] = oacc[t][i] * inv;
        }
    }
}

// ---------------------------------------------------------------------------
extern "C" void kernel_launch(void* const* d_in, const int* in_sizes, int n_in,
                              void* d_out, int out_size, void* d_ws, size_t ws_size,
                              hipStream_t stream) {
    const float* x  = (const float*)d_in[0];
    const float* y  = (const float*)d_in[1];
    const float* Wq = (const float*)d_in[2];
    const float* bq = (const float*)d_in[3];
    const float* Wk = (const float*)d_in[4];
    const float* bk = (const float*)d_in[5];
    const float* Wv = (const float*)d_in[6];
    const float* bv = (const float*)d_in[7];
    float* out = (float*)d_out;

    _Float16* Qh = (_Float16*)d_ws;                    // 16*4096*64 halves = 8 MB
    _Float16* Kh = Qh + (long)NPAIR * NTOK * DHEAD;    // 8 MB
    _Float16* Vh = Kh + (long)NPAIR * NTOK * DHEAD;    // 8 MB

    proj_kernel<<<dim3(BATCH * NTOK / 64), dim3(256), 0, stream>>>(
        x, y, Wq, bq, Wk, bk, Wv, bv, Qh, Kh, Vh);

    attn_kernel<<<dim3(NTOK / 64, NPAIR), dim3(256), 0, stream>>>(Qh, Kh, Vh, out);
}
